// Attention_68564857914003
// MI455X (gfx1250) — matrix operand order repaired and checked by reading, not compiled, not run on hardware
//
#include <hip/hip_runtime.h>
#include <hip/hip_bf16.h>

typedef __attribute__((ext_vector_type(16))) _Float16 v16h;
typedef __attribute__((ext_vector_type(8)))  _Float16 v8h;
typedef __attribute__((ext_vector_type(8)))  float    v8f;

#define WMMA_F16(a, b, c) \
  __builtin_amdgcn_wmma_f32_16x16x32_f16(false, (a), false, (b), (short)0, (c), false, false)

static constexpr int Bn = 4, Sn = 2048, Dn = 1024, Hn = 16, HDn = 64;
static constexpr int Mn = Bn * Sn;

// ---- CDNA5 async global->LDS staging (ASYNCcnt path), with safe fallback ----
#if __has_builtin(__builtin_amdgcn_global_load_async_to_lds_b128) && \
    __has_builtin(__builtin_amdgcn_s_wait_asynccnt)
#define HAVE_ASYNC 1
#else
#define HAVE_ASYNC 0
#endif

// The builtin's params are int4* in global (addrspace 1) / LDS (addrspace 3).
typedef int v4i __attribute__((vector_size(16)));
typedef __attribute__((address_space(1))) v4i g_v4i;
typedef __attribute__((address_space(3))) v4i l_v4i;

// Move 16 bytes global->LDS. Async path: no VGPR round-trip, tracked by ASYNCcnt.
static __device__ __forceinline__ void stage16(_Float16* l, const _Float16* g) {
#if HAVE_ASYNC
  __builtin_amdgcn_global_load_async_to_lds_b128(
      (g_v4i*)(uintptr_t)g,
      (l_v4i*)(unsigned)(uintptr_t)l,   // generic LDS ptr low 32 bits = LDS offset
      0, 0);
#else
  *(v8h*)l = *(const v8h*)g;
#endif
}
static __device__ __forceinline__ void stage_wait() {
#if HAVE_ASYNC
  __builtin_amdgcn_s_wait_asynccnt(0);
#endif
}

static __device__ __forceinline__ v16h cat8(v8h lo, v8h hi) {
  return __builtin_shufflevector(lo, hi, 0, 1, 2, 3, 4, 5, 6, 7,
                                         8, 9, 10, 11, 12, 13, 14, 15);
}

// A-matrix fragment (16 rows x 32 K, f16). Row = lane&15; lane group g = lane>>4
// holds K = {8g..8g+7} U {8g+16..8g+23} (ISA 7.12.2, 16-bit A 16x32).
static __device__ __forceinline__ v16h load_frag_a(const _Float16* t, int ld, int lane) {
  const int g = lane >> 4, r = lane & 15;
  const _Float16* p = t + r * ld + g * 8;
  v8h lo = *(const v8h*)p;
  v8h hi = *(const v8h*)(p + 16);
  return cat8(lo, hi);
}

// B fragment (32K x 16N) from a tile stored [n][k] (K contiguous per n-row):
// column n = lane&15, lane group g holds K = {8g..8g+7} U {8g+16..8g+23}.
static __device__ __forceinline__ v16h load_frag_b_nk(const _Float16* t, int ld, int lane) {
  const int g = lane >> 4, n = lane & 15;
  const _Float16* p = t + n * ld + g * 8;
  v8h lo = *(const v8h*)p;
  v8h hi = *(const v8h*)(p + 16);
  return cat8(lo, hi);
}

// ---------------------------------------------------------------- cast fp32 -> f16
__global__ __launch_bounds__(256) void cast_f32_f16(const float* __restrict__ x,
                                                    _Float16* __restrict__ y, int n) {
  int i = blockIdx.x * 256 + threadIdx.x;
  if (i < n) y[i] = (_Float16)x[i];
}

// ---------------------------------------------------------------- GEMM: Y = X * W^T (+bias)
// X: [M,K] f16 row-major; W: [N,K] f16 row-major; Y: [M,N] f32.
// 128x128 block tile, 8 waves (64x32 each), K-step 32, async-staged LDS tiles.
template <bool HAS_BIAS>
__global__ __launch_bounds__(256) void gemm_xwT(const _Float16* __restrict__ X,
                                                const _Float16* __restrict__ W,
                                                const float* __restrict__ bias,
                                                float* __restrict__ Y,
                                                int M, int N, int K) {
  __shared__ __align__(16) _Float16 As[128 * 32];
  __shared__ __align__(16) _Float16 Bs[128 * 32];
  const int tid = threadIdx.x, lane = tid & 31, wave = tid >> 5;
  const int m0 = blockIdx.y * 128, n0 = blockIdx.x * 128;
  const int wm = (wave & 1) * 64, wn = (wave >> 1) * 32;

  const v8f vzero = {};
  v8f acc[4][2];
#pragma unroll
  for (int i = 0; i < 4; ++i) { acc[i][0] = vzero; acc[i][1] = vzero; }

  const int crow = tid >> 1;          // 0..127: tile row this thread stages
  const int coff = (tid & 1) * 16;    // halves within the 32-wide K slab
  const _Float16* ga = X + (size_t)(m0 + crow) * K + coff;
  const _Float16* gb = W + (size_t)(n0 + crow) * K + coff;
  _Float16* la = &As[crow * 32 + coff];
  _Float16* lb = &Bs[crow * 32 + coff];

  for (int k0 = 0; k0 < K; k0 += 32) {
    stage16(la,     ga + k0);
    stage16(la + 8, ga + k0 + 8);
    stage16(lb,     gb + k0);
    stage16(lb + 8, gb + k0 + 8);
    if (k0 + 32 < K) {                       // pull next K slab toward L2/WGP$
      __builtin_prefetch(ga + k0 + 32, 0, 1);
      __builtin_prefetch(gb + k0 + 32, 0, 1);
    }
    stage_wait();
    __syncthreads();
    v16h bf[2];
#pragma unroll
    for (int nt = 0; nt < 2; ++nt)
      bf[nt] = load_frag_b_nk(&Bs[(wn + nt * 16) * 32], 32, lane);
#pragma unroll
    for (int mt = 0; mt < 4; ++mt) {
      v16h af = load_frag_a(&As[(wm + mt * 16) * 32], 32, lane);
      acc[mt][0] = WMMA_F16(af, bf[0], acc[mt][0]);
      acc[mt][1] = WMMA_F16(af, bf[1], acc[mt][1]);
    }
    __syncthreads();
  }

  const int g = lane >> 4, nloc = lane & 15;
#pragma unroll
  for (int mt = 0; mt < 4; ++mt)
#pragma unroll
    for (int nt = 0; nt < 2; ++nt) {
      const int n = n0 + wn + nt * 16 + nloc;
      const float bv = HAS_BIAS ? bias[n] : 0.0f;
#pragma unroll
      for (int r = 0; r < 8; ++r) {
        const int m = m0 + wm + mt * 16 + g * 8 + r;   // C layout: lane half picks M 0-7 / 8-15
        Y[(size_t)m * N + n] = acc[mt][nt][r] + bv;
      }
    }
}

// ---------------------------------------------------------------- RoPE: Q,K -> f16 [B,H,S,hd]
__global__ __launch_bounds__(256) void rope_pack(const float* __restrict__ Q,
                                                 const float* __restrict__ Km,
                                                 const int* __restrict__ pos_ids,
                                                 _Float16* __restrict__ Qh,
                                                 _Float16* __restrict__ Kh) {
  const int t = blockIdx.x * 256 + threadIdx.x;  // B*S*H*32 threads
  const int i = t & 31;                // rotary pair index (d, d+32)
  const int h = (t >> 5) & (Hn - 1);
  const int s = (t >> 9) & (Sn - 1);
  const int b = t >> 20;
  const float pos = (float)pos_ids[b * Sn + s];
  const float inv = __powf(10000.0f, -(float)i * (1.0f / 32.0f));
  float sn, c;
  __sincosf(pos * inv, &sn, &c);
  const size_t src = ((size_t)(b * Sn + s)) * Dn + h * HDn;
  const size_t dst = (((size_t)b * Hn + h) * Sn + s) * HDn;
  const float q1 = Q[src + i], q2 = Q[src + i + 32];
  const float k1 = Km[src + i], k2 = Km[src + i + 32];
  const float scale = 0.125f;          // 1/sqrt(64) folded into Q
  Qh[dst + i]      = (_Float16)((q1 * c - q2 * sn) * scale);
  Qh[dst + i + 32] = (_Float16)((q2 * c + q1 * sn) * scale);
  Kh[dst + i]      = (_Float16)(k1 * c - k2 * sn);
  Kh[dst + i + 32] = (_Float16)(k2 * c + k1 * sn);
}

// ---------------------------------------------------------------- V: fp32 [B,S,D] -> f16 [B,H,hd,S]
// LDS-tiled 64x64 transpose so both global sides stay coalesced; the transposed
// layout makes flash-attention's V B-fragments contiguous ds_load_b128s.
__global__ __launch_bounds__(256) void transpose_v(const float* __restrict__ V,
                                                   _Float16* __restrict__ Vt) {
  __shared__ _Float16 T[64][65];       // +1 pad: conflict-free column reads
  const int nsb = Sn / 64;
  const int s0 = (blockIdx.x % nsb) * 64;
  const int h  = (blockIdx.x / nsb) % Hn;
  const int b  = blockIdx.x / (nsb * Hn);
  const int r  = threadIdx.x >> 2;     // 0..63
  const int c0 = (threadIdx.x & 3) * 16;
  const float* src = V + ((size_t)(b * Sn + s0 + r)) * Dn + h * HDn + c0;
#pragma unroll
  for (int j = 0; j < 16; ++j) T[r][c0 + j] = (_Float16)src[j];
  __syncthreads();
  _Float16* dst = Vt + (((size_t)b * Hn + h) * HDn + r) * Sn + s0 + c0;
#pragma unroll
  for (int j = 0; j < 16; ++j) dst[j] = T[c0 + j][r];
}

// ---------------------------------------------------------------- flash attention
// 1 block = 4 waves = 64 query rows of one (b,h). K tiles [key][hd], V tiles
// [hd][key] (pre-transposed) async-staged through LDS; online softmax in regs.
__global__ __launch_bounds__(128) void flash_attn(const _Float16* __restrict__ Qh,
                                                  const _Float16* __restrict__ Kh,
                                                  const _Float16* __restrict__ Vt,
                                                  _Float16* __restrict__ Oh) {
  __shared__ __align__(16) _Float16 Ks[64 * 64];      // [key][hd]
  __shared__ __align__(16) _Float16 Vs[64 * 64];      // [hd][key]
  __shared__ __align__(16) _Float16 Ps[4][16 * 64];   // per-wave P staging (C->A layout)
  const int tid = threadIdx.x, lane = tid & 31, wave = tid >> 5;
  const int nqb = Sn / 64;
  const int q0 = (blockIdx.x % nqb) * 64;
  const int h  = (blockIdx.x / nqb) % Hn;
  const int b  = blockIdx.x / (nqb * Hn);
  const size_t bh  = ((size_t)b * Hn + h) * Sn * HDn;   // Q/K base
  const size_t bhv = ((size_t)b * Hn + h) * HDn * Sn;   // Vt base
  const int qrow = q0 + wave * 16;

  const v16h qa0 = load_frag_a(Qh + bh + (size_t)qrow * HDn, HDn, lane);
  const v16h qa1 = load_frag_a(Qh + bh + (size_t)qrow * HDn + 32, HDn, lane);

  const v8f vzero = {};
  v8f o[4];
  float mrow[8], lrow[8];
#pragma unroll
  for (int dt = 0; dt < 4; ++dt) o[dt] = vzero;
#pragma unroll
  for (int r = 0; r < 8; ++r) { mrow[r] = -1e30f; lrow[r] = 0.0f; }

  const int crow = tid >> 1;           // 0..63: tile row staged by this thread
  const int coff = (tid & 1) * 32;

  for (int s0 = 0; s0 < Sn; s0 += 64) {
    const _Float16* gk = Kh + bh  + (size_t)(s0 + crow) * HDn + coff;  // K row = key
    const _Float16* gv = Vt + bhv + (size_t)crow * Sn + s0 + coff;     // Vt row = hd
#pragma unroll
    for (int j = 0; j < 4; ++j) {
      stage16(&Ks[crow * 64 + coff + j * 8], gk + j * 8);
      stage16(&Vs[crow * 64 + coff + j * 8], gv + j * 8);
    }
    stage_wait();
    __syncthreads();

    // scores S = Q * K^T  (Q pre-scaled by 1/sqrt(hd))
    float sc[4][8];
#pragma unroll
    for (int nt = 0; nt < 4; ++nt) {
      v16h kb0 = load_frag_b_nk(&Ks[nt * 16 * 64], 64, lane);
      v16h kb1 = load_frag_b_nk(&Ks[nt * 16 * 64 + 32], 64, lane);
      v8f sacc = {};
      sacc = WMMA_F16(qa0, kb0, sacc);
      sacc = WMMA_F16(qa1, kb1, sacc);
#pragma unroll
      for (int r = 0; r < 8; ++r) sc[nt][r] = sacc[r];
    }

    // online softmax; each row lives entirely in one 16-lane half (C layout),
    // so xor-shuffles with masks 1,2,4,8 reduce across the row's 16 columns.
    float corr[8];
#pragma unroll
    for (int r = 0; r < 8; ++r) {
      float cm = fmaxf(fmaxf(sc[0][r], sc[1][r]), fmaxf(sc[2][r], sc[3][r]));
#pragma unroll
      for (int msk = 1; msk <= 8; msk <<= 1) cm = fmaxf(cm, __shfl_xor(cm, msk, 32));
      const float mnew = fmaxf(mrow[r], cm);
      corr[r] = __expf(mrow[r] - mnew);
      float rs = 0.0f;
#pragma unroll
      for (int nt = 0; nt < 4; ++nt) { sc[nt][r] = __expf(sc[nt][r] - mnew); rs += sc[nt][r]; }
#pragma unroll
      for (int msk = 1; msk <= 8; msk <<= 1) rs += __shfl_xor(rs, msk, 32);
      lrow[r] = lrow[r] * corr[r] + rs;
      mrow[r] = mnew;
    }
#pragma unroll
    for (int dt = 0; dt < 4; ++dt)
#pragma unroll
      for (int r = 0; r < 8; ++r) o[dt][r] *= corr[r];

    // P: C-layout regs -> per-wave LDS patch -> A-layout fragments (same-wave
    // DS ordering; compiler inserts s_wait_dscnt).
    {
      const int g = lane >> 4, nloc = lane & 15;
      _Float16* ps = Ps[wave];
#pragma unroll
      for (int nt = 0; nt < 4; ++nt)
#pragma unroll
        for (int r = 0; r < 8; ++r)
          ps[(g * 8 + r) * 64 + nt * 16 + nloc] = (_Float16)sc[nt][r];
      v16h pa0 = load_frag_a(ps, 64, lane);
      v16h pa1 = load_frag_a(ps + 32, 64, lane);
#pragma unroll
      for (int dt = 0; dt < 4; ++dt) {
        v16h vb0 = load_frag_b_nk(&Vs[dt * 16 * 64], 64, lane);       // keys 0..31
        v16h vb1 = load_frag_b_nk(&Vs[dt * 16 * 64 + 32], 64, lane);  // keys 32..63
        o[dt] = WMMA_F16(pa0, vb0, o[dt]);
        o[dt] = WMMA_F16(pa1, vb1, o[dt]);
      }
    }
    __syncthreads();
  }

  const int g = lane >> 4, nloc = lane & 15;
#pragma unroll
  for (int dt = 0; dt < 4; ++dt)
#pragma unroll
    for (int r = 0; r < 8; ++r) {
      const int m = qrow + g * 8 + r;
      const int d = h * HDn + dt * 16 + nloc;
      Oh[((size_t)b * Sn + m) * Dn + d] = (_Float16)(o[dt][r] / lrow[r]);
    }
}

// ---------------------------------------------------------------- launch
extern "C" void kernel_launch(void* const* d_in, const int* in_sizes, int n_in,
                              void* d_out, int out_size, void* d_ws, size_t ws_size,
                              hipStream_t stream) {
  (void)in_sizes; (void)n_in; (void)out_size; (void)ws_size;
  const float* hs  = (const float*)d_in[0];
  const int*   pid = (const int*)d_in[1];
  const float* Wq  = (const float*)d_in[2];
  const float* bq  = (const float*)d_in[3];
  const float* Wk  = (const float*)d_in[4];
  const float* bk  = (const float*)d_in[5];
  const float* Wv  = (const float*)d_in[6];
  const float* bv  = (const float*)d_in[7];
  const float* Wo  = (const float*)d_in[8];
  float* out = (float*)d_out;

  char* w = (char*)d_ws;
  size_t off = 0;
  auto alloc = [&](size_t bytes) -> void* {
    void* p = w + off;
    off = (off + bytes + 255) & ~(size_t)255;
    return p;
  };
  _Float16* xh  = (_Float16*)alloc((size_t)Mn * Dn * 2);
  _Float16* wqh = (_Float16*)alloc((size_t)Dn * Dn * 2);
  _Float16* wkh = (_Float16*)alloc((size_t)Dn * Dn * 2);
  _Float16* wvh = (_Float16*)alloc((size_t)Dn * Dn * 2);
  _Float16* woh = (_Float16*)alloc((size_t)Dn * Dn * 2);
  float*    qf  = (float*)alloc((size_t)Mn * Dn * 4);
  float*    kf  = (float*)alloc((size_t)Mn * Dn * 4);
  float*    vf  = (float*)alloc((size_t)Mn * Dn * 4);
  _Float16* qh  = (_Float16*)alloc((size_t)Mn * Dn * 2);
  _Float16* kh  = (_Float16*)alloc((size_t)Mn * Dn * 2);
  _Float16* vh  = (_Float16*)alloc((size_t)Mn * Dn * 2);  // [B,H,hd,S]
  _Float16* ah  = (_Float16*)qf;   // qf is dead after rope_pack; reuse for attn output

  cast_f32_f16<<<(Mn * Dn) / 256, 256, 0, stream>>>(hs, xh, Mn * Dn);
  cast_f32_f16<<<(Dn * Dn) / 256, 256, 0, stream>>>(Wq, wqh, Dn * Dn);
  cast_f32_f16<<<(Dn * Dn) / 256, 256, 0, stream>>>(Wk, wkh, Dn * Dn);
  cast_f32_f16<<<(Dn * Dn) / 256, 256, 0, stream>>>(Wv, wvh, Dn * Dn);
  cast_f32_f16<<<(Dn * Dn) / 256, 256, 0, stream>>>(Wo, woh, Dn * Dn);

  dim3 gg(Dn / 128, Mn / 128);
  gemm_xwT<true><<<gg, 256, 0, stream>>>(xh, wqh, bq, qf, Mn, Dn, Dn);
  gemm_xwT<true><<<gg, 256, 0, stream>>>(xh, wkh, bk, kf, Mn, Dn, Dn);
  gemm_xwT<true><<<gg, 256, 0, stream>>>(xh, wvh, bv, vf, Mn, Dn, Dn);

  rope_pack<<<(Bn * Sn * Hn * 32) / 256, 256, 0, stream>>>(qf, kf, pid, qh, kh);
  transpose_v<<<Bn * Hn * (Sn / 64), 256, 0, stream>>>(vf, vh);

  flash_attn<<<Bn * Hn * (Sn / 64), 128, 0, stream>>>(qh, kh, vh, ah);

  gemm_xwT<false><<<gg, 256, 0, stream>>>(ah, woh, nullptr, out, Mn, Dn, Dn);
}
